// GraphUNet_87574383165971
// MI455X (gfx1250) — matrix operand order repaired and checked by reading, not compiled, not run on hardware
//
#include <hip/hip_runtime.h>
#include <hip/hip_bf16.h>
#include <math.h>

typedef __bf16 bf16;
typedef __attribute__((ext_vector_type(16))) __bf16 v16bf;
typedef __attribute__((ext_vector_type(8)))  __bf16 v8bf;
typedef __attribute__((ext_vector_type(8)))  float  v8f;

#define THREADS 256
static inline unsigned gblocks(size_t n) { return (unsigned)((n + THREADS - 1) / THREADS); }

// ---------------------------------------------------------------------------
// WMMA GEMM:  C[M x N] = A[M x K] * Bt[N x K]^T   (bf16 in, f32 accum/out)
// One wave (32 threads) computes a 16x64 output strip (4 WMMA tiles sharing
// the A fragment).  Requires: M%16==0, N%64==0, K%32==0, lda%8==0, ldbt%16==0.
// A/B fragments use contiguous K order (lane half h: k = 16h..16h+15), not the ISA table order.
// ---------------------------------------------------------------------------
__global__ void __launch_bounds__(32)
k_wmma_gemm_bt(const bf16* __restrict__ A, const bf16* __restrict__ Bt,
               float* __restrict__ C, int M, int N, int K,
               int lda, int ldbt, int ldc)
{
    (void)M; (void)N;
    const int lane = threadIdx.x;
    const int half = lane >> 4;     // 0: lanes 0-15, 1: lanes 16-31
    const int lr   = lane & 15;
    const int tm   = blockIdx.y * 16;
    const int tn   = blockIdx.x * 64;

    // A fragment: lane<16 holds row tm+lr, K = kk..kk+15
    //             lane>=16 holds row tm+lr, K = kk+16..kk+31
    const bf16* ap  = A  + (size_t)(tm + lr) * lda + half * 16;
    // B fragment: lane<16 holds col tn+lr, K = kk..kk+15 ; lane>=16 K = kk+16..kk+31
    const bf16* bp0 = Bt + (size_t)(tn + lr) * ldbt + half * 16;
    const size_t bstep = (size_t)16 * ldbt;

    v8f acc0 = {}; v8f acc1 = {}; v8f acc2 = {}; v8f acc3 = {};

    for (int kk = 0; kk < K; kk += 32) {
        union { v16bf v; v8bf h[2]; } ua;
        ua.h[0] = *(const v8bf*)(ap + kk);
        ua.h[1] = *(const v8bf*)(ap + kk + 8);
        v16bf af = ua.v;
        v16bf b0 = *(const v16bf*)(bp0 + kk);
        v16bf b1 = *(const v16bf*)(bp0 + bstep + kk);
        v16bf b2 = *(const v16bf*)(bp0 + 2 * bstep + kk);
        v16bf b3 = *(const v16bf*)(bp0 + 3 * bstep + kk);
        acc0 = __builtin_amdgcn_wmma_f32_16x16x32_bf16(false, af, false, b0, (short)0, acc0, false, false);
        acc1 = __builtin_amdgcn_wmma_f32_16x16x32_bf16(false, af, false, b1, (short)0, acc1, false, false);
        acc2 = __builtin_amdgcn_wmma_f32_16x16x32_bf16(false, af, false, b2, (short)0, acc2, false, false);
        acc3 = __builtin_amdgcn_wmma_f32_16x16x32_bf16(false, af, false, b3, (short)0, acc3, false, false);
    }

    // C/D layout: VGPR v -> row tm + half*8 + v, col tn + lr (+16*tile)
    float* cb = C + (size_t)(tm + half * 8) * ldc + tn + lr;
#pragma unroll
    for (int v = 0; v < 8; ++v) {
        float* cr = cb + (size_t)v * ldc;
        cr[0]  = acc0[v];
        cr[16] = acc1[v];
        cr[32] = acc2[v];
        cr[48] = acc3[v];
    }
}

// ---------------------------------------------------------------------------
// Support kernels
// ---------------------------------------------------------------------------
__global__ void k_scatter_edges(const int* __restrict__ ei, float* __restrict__ A,
                                int E, int n)
{
    int t = blockIdx.x * blockDim.x + threadIdx.x;
    if (t >= E) return;
    int s = ei[t], d = ei[E + t];
    atomicAdd(&A[(size_t)s * n + d], 1.0f);
}

__global__ void k_rowsum_rsqrt(const float* __restrict__ A, float* __restrict__ d, int n)
{
    __shared__ float sm[THREADS];
    int row = blockIdx.x;
    const float* ar = A + (size_t)row * n;
    float s = 0.0f;
    for (int j = threadIdx.x; j < n; j += THREADS) s += ar[j];
    sm[threadIdx.x] = s;
    __syncthreads();
    for (int st = THREADS / 2; st > 0; st >>= 1) {
        if (threadIdx.x < st) sm[threadIdx.x] += sm[threadIdx.x + st];
        __syncthreads();
    }
    if (threadIdx.x == 0) d[row] = rsqrtf(sm[0] + 2.0f);   // improved: +2 self loop
}

__global__ void k_cvt_bf16(const float* __restrict__ src, bf16* __restrict__ dst,
                           unsigned long long n)
{
    unsigned long long i = (unsigned long long)blockIdx.x * blockDim.x + threadIdx.x;
    if (i >= n) return;
    dst[i] = (bf16)src[i];
}

__global__ void k_transpose_w(const float* __restrict__ W, bf16* __restrict__ WT,
                              int rows, int cols)   // W: rows x cols -> WT: cols x rows
{
    int i = blockIdx.x * blockDim.x + threadIdx.x;
    if (i >= rows * cols) return;
    int r = i / cols, c = i % cols;
    WT[(size_t)c * rows + r] = (bf16)W[i];
}

__global__ void k_scale_rows_T(const float* __restrict__ XW, const float* __restrict__ d,
                               bf16* __restrict__ yT, int n, int c)  // yT: c x n
{
    int i = blockIdx.x * blockDim.x + threadIdx.x;
    if (i >= n * c) return;
    int row = i / c, col = i % c;
    yT[(size_t)col * n + row] = (bf16)(d[row] * XW[i]);
}

__global__ void k_epilogue(const float* __restrict__ Z, const float* __restrict__ XW,
                           const float* __restrict__ d, const float* __restrict__ b,
                           float* __restrict__ out, int n, int c, int relu)
{
    int i = blockIdx.x * blockDim.x + threadIdx.x;
    if (i >= n * c) return;
    int row = i / c, col = i % c;
    float di = d[row];
    float v = di * Z[i] + 2.0f * di * di * XW[i] + b[col];
    out[i] = relu ? fmaxf(v, 0.0f) : v;
}

__global__ void k_pnorm(const float* __restrict__ p, float* __restrict__ out, int c)
{
    __shared__ float sm[THREADS];
    float s = 0.0f;
    for (int j = threadIdx.x; j < c; j += THREADS) s += p[j] * p[j];
    sm[threadIdx.x] = s;
    __syncthreads();
    for (int st = THREADS / 2; st > 0; st >>= 1) {
        if (threadIdx.x < st) sm[threadIdx.x] += sm[threadIdx.x + st];
        __syncthreads();
    }
    if (threadIdx.x == 0) out[0] = sqrtf(sm[0]);
}

__global__ void k_score(const float* __restrict__ x, const float* __restrict__ p,
                        const float* __restrict__ pn, float* __restrict__ score,
                        int n, int c)
{
    __shared__ float sm[THREADS];
    int row = blockIdx.x;
    const float* xr = x + (size_t)row * c;
    float s = 0.0f;
    for (int j = threadIdx.x; j < c; j += THREADS) s += xr[j] * p[j];
    sm[threadIdx.x] = s;
    __syncthreads();
    for (int st = THREADS / 2; st > 0; st >>= 1) {
        if (threadIdx.x < st) sm[threadIdx.x] += sm[threadIdx.x + st];
        __syncthreads();
    }
    if (threadIdx.x == 0) score[row] = tanhf(sm[0] / pn[0]);
}

__global__ void k_keyinit(const float* __restrict__ s,
                          unsigned long long* __restrict__ keys, int n)
{
    int i = blockIdx.x * blockDim.x + threadIdx.x;
    if (i >= n) return;
    unsigned u = __float_as_uint(s[i]);
    u ^= (u >> 31) ? 0xFFFFFFFFu : 0x80000000u;   // monotone map (ascending)
    u = ~u;                                        // descending score
    keys[i] = ((unsigned long long)u << 32) | (unsigned)i;  // ties: lower idx first
}

__global__ void k_bitonic(unsigned long long* __restrict__ keys, int j, int kk)
{
    int i = blockIdx.x * blockDim.x + threadIdx.x;
    int l = i ^ j;
    if (l <= i) return;
    unsigned long long a = keys[i], b = keys[l];
    bool up = ((i & kk) == 0);
    if (up ? (a > b) : (a < b)) { keys[i] = b; keys[l] = a; }
}

__global__ void k_permext(const unsigned long long* __restrict__ keys,
                          int* __restrict__ perm, int k)
{
    int i = blockIdx.x * blockDim.x + threadIdx.x;
    if (i >= k) return;
    perm[i] = (int)(keys[i] & 0xFFFFFFFFull);
}

__global__ void k_gather_x(const float* __restrict__ xin, const float* __restrict__ score,
                           const int* __restrict__ perm, float* __restrict__ xout,
                           int k, int c)
{
    int i = blockIdx.x * blockDim.x + threadIdx.x;
    if (i >= k * c) return;
    int r = i / c, j = i % c;
    int pi = perm[r];
    xout[i] = xin[(size_t)pi * c + j] * score[pi];
}

// Ga[i][t] = As[perm[i]][t] with As = A(diag forced to 1)   (row gather)
__global__ void k_ga(const float* __restrict__ A, const int* __restrict__ perm,
                     bf16* __restrict__ Ga, int k, int n)
{
    unsigned long long idx = (unsigned long long)blockIdx.x * blockDim.x + threadIdx.x;
    if (idx >= (unsigned long long)k * n) return;
    int i = (int)(idx / n), t = (int)(idx % n);
    int pi = perm[i];
    float v = (t == pi) ? 1.0f : A[(size_t)pi * n + t];
    Ga[idx] = (bf16)v;
}

// Gb[j][t] = As[t][perm[j]]   (column gather, stored row-major => acts as Bt)
__global__ void k_gb(const float* __restrict__ A, const int* __restrict__ perm,
                     bf16* __restrict__ Gb, int k, int n)
{
    unsigned long long idx = (unsigned long long)blockIdx.x * blockDim.x + threadIdx.x;
    if (idx >= (unsigned long long)k * n) return;
    int j = (int)(idx / n), t = (int)(idx % n);
    int pj = perm[j];
    float v = (t == pj) ? 1.0f : A[(size_t)t * n + pj];
    Gb[idx] = (bf16)v;
}

__global__ void k_zero_diag(float* __restrict__ A, int k)
{
    int i = blockIdx.x * blockDim.x + threadIdx.x;
    if (i >= k) return;
    A[(size_t)i * k + i] = 0.0f;
}

__global__ void k_scatter_add(const float* __restrict__ src, const int* __restrict__ perm,
                              float* __restrict__ dst, int k, int c)
{
    int i = blockIdx.x * blockDim.x + threadIdx.x;
    if (i >= k * c) return;
    int r = i / c, j = i % c;
    dst[(size_t)perm[r] * c + j] += src[i];   // perm rows are unique
}

// ---------------------------------------------------------------------------
// Host-side orchestration
// ---------------------------------------------------------------------------
struct Ws {
    float *A0, *A1, *A2, *A3;
    bf16  *Abf, *Ga, *Gb;      // Ga/Gb alias the Abf region (phases are disjoint)
    float *x0, *x1, *x2, *xa, *xb, *XW, *Zf;
    bf16  *xbf, *yT, *WT;
    float *dvec, *score, *pnorm;
    unsigned long long *keys;
    int *perm0, *perm1, *perm2;
};

static void run_gcn(hipStream_t st, Ws& w, const float* Af, int n,
                    const float* xin, int cin, const float* W, const float* bvec,
                    int cout, int relu, float* xout)
{
    k_rowsum_rsqrt<<<n, THREADS, 0, st>>>(Af, w.dvec, n);
    k_transpose_w<<<gblocks((size_t)cin * cout), THREADS, 0, st>>>(W, w.WT, cin, cout);
    k_cvt_bf16<<<gblocks((size_t)n * cin), THREADS, 0, st>>>(xin, w.xbf, (unsigned long long)n * cin);
    {   // XW = x @ W
        dim3 g(cout / 64, n / 16);
        k_wmma_gemm_bt<<<g, 32, 0, st>>>(w.xbf, w.WT, w.XW, n, cout, cin, cin, cin, cout);
    }
    k_scale_rows_T<<<gblocks((size_t)n * cout), THREADS, 0, st>>>(w.XW, w.dvec, w.yT, n, cout);
    k_cvt_bf16<<<gblocks((size_t)n * n), THREADS, 0, st>>>(Af, w.Abf, (unsigned long long)n * n);
    {   // Z = A @ (d .* XW)
        dim3 g(cout / 64, n / 16);
        k_wmma_gemm_bt<<<g, 32, 0, st>>>(w.Abf, w.yT, w.Zf, n, cout, n, n, n, cout);
    }
    k_epilogue<<<gblocks((size_t)n * cout), THREADS, 0, st>>>(w.Zf, w.XW, w.dvec, bvec, xout, n, cout, relu);
}

static void run_pool(hipStream_t st, Ws& w, const float* Af, int n,
                     const float* xin, const float* p, int k,
                     float* xout, float* Aout, int* perm)
{
    k_pnorm<<<1, THREADS, 0, st>>>(p, w.pnorm, 256);
    k_score<<<n, THREADS, 0, st>>>(xin, p, w.pnorm, w.score, n, 256);
    k_keyinit<<<gblocks((size_t)n), THREADS, 0, st>>>(w.score, w.keys, n);
    for (int kk = 2; kk <= n; kk <<= 1)
        for (int j = kk >> 1; j > 0; j >>= 1)
            k_bitonic<<<n / THREADS, THREADS, 0, st>>>(w.keys, j, kk);
    k_permext<<<gblocks((size_t)k), THREADS, 0, st>>>(w.keys, perm, k);
    k_gather_x<<<gblocks((size_t)k * 256), THREADS, 0, st>>>(xin, w.score, perm, xout, k, 256);
    k_ga<<<gblocks((size_t)k * n), THREADS, 0, st>>>(Af, perm, w.Ga, k, n);
    k_gb<<<gblocks((size_t)k * n), THREADS, 0, st>>>(Af, perm, w.Gb, k, n);
    {   // Ak = (As @ As)[perm][:,perm]  (gathered k x k x n GEMM)
        dim3 g(k / 64, k / 16);
        k_wmma_gemm_bt<<<g, 32, 0, st>>>(w.Ga, w.Gb, Aout, k, k, n, n, n, k);
    }
    k_zero_diag<<<gblocks((size_t)k), THREADS, 0, st>>>(Aout, k);   // remove self loops
}

extern "C" void kernel_launch(void* const* d_in, const int* in_sizes, int n_in,
                              void* d_out, int out_size, void* d_ws, size_t ws_size,
                              hipStream_t stream)
{
    (void)n_in; (void)out_size; (void)ws_size;

    const float* x   = (const float*)d_in[0];
    const int*   ei  = (const int*)d_in[1];
    const float* wd0 = (const float*)d_in[2];  const float* bd0 = (const float*)d_in[3];
    const float* wd1 = (const float*)d_in[4];  const float* bd1 = (const float*)d_in[5];
    const float* wd2 = (const float*)d_in[6];  const float* bd2 = (const float*)d_in[7];
    const float* wd3 = (const float*)d_in[8];  const float* bd3 = (const float*)d_in[9];
    const float* p0  = (const float*)d_in[10];
    const float* p1  = (const float*)d_in[11];
    const float* p2  = (const float*)d_in[12];
    const float* wu0 = (const float*)d_in[13]; const float* bu0 = (const float*)d_in[14];
    const float* wu1 = (const float*)d_in[15]; const float* bu1 = (const float*)d_in[16];
    const float* wu2 = (const float*)d_in[17]; const float* bu2 = (const float*)d_in[18];

    const int N0 = 4096, N1 = 2048, N2 = 1024, N3 = 512;
    const int E  = in_sizes[1] / 2;

    // Bump-allocate workspace (deterministic layout every call)
    char* base = (char*)d_ws;
    size_t off = 0;
    auto alloc = [&](size_t bytes) -> char* {
        char* ptr = base + off;
        off = (off + bytes + 255) & ~(size_t)255;
        return ptr;
    };

    Ws w;
    w.A0   = (float*)alloc((size_t)N0 * N0 * 4);
    w.A1   = (float*)alloc((size_t)N1 * N1 * 4);
    w.A2   = (float*)alloc((size_t)N2 * N2 * 4);
    w.A3   = (float*)alloc((size_t)N3 * N3 * 4);
    w.Abf  = (bf16*)alloc((size_t)N0 * N0 * 2);          // 32MB region
    w.Ga   = w.Abf;                                      // alias (max 16MB)
    w.Gb   = w.Abf + (size_t)N1 * N0;                    // alias at +16MB
    w.x0   = (float*)alloc((size_t)N0 * 256 * 4);
    w.x1   = (float*)alloc((size_t)N1 * 256 * 4);
    w.x2   = (float*)alloc((size_t)N2 * 256 * 4);
    w.xa   = (float*)alloc((size_t)N0 * 256 * 4);
    w.xb   = (float*)alloc((size_t)N0 * 256 * 4);
    w.XW   = (float*)alloc((size_t)N0 * 256 * 4);
    w.Zf   = (float*)alloc((size_t)N0 * 256 * 4);
    w.xbf  = (bf16*)alloc((size_t)N0 * 256 * 2);
    w.yT   = (bf16*)alloc((size_t)256 * N0 * 2);
    w.WT   = (bf16*)alloc((size_t)256 * 256 * 2);
    w.dvec = (float*)alloc((size_t)N0 * 4);
    w.score= (float*)alloc((size_t)N0 * 4);
    w.pnorm= (float*)alloc(256);
    w.keys = (unsigned long long*)alloc((size_t)N0 * 8);
    w.perm0= (int*)alloc((size_t)N1 * 4);
    w.perm1= (int*)alloc((size_t)N2 * 4);
    w.perm2= (int*)alloc((size_t)N3 * 4);

    // ---- Build dense adjacency A0 from edge_index ----
    hipMemsetAsync(w.A0, 0, (size_t)N0 * N0 * 4, stream);
    k_scatter_edges<<<gblocks((size_t)E), THREADS, 0, stream>>>(ei, w.A0, E, N0);

    // ---- Down path ----
    run_gcn(stream, w, w.A0, N0, x, 128, wd0, bd0, 256, 1, w.x0);          // xs[0]

    run_pool(stream, w, w.A0, N0, w.x0, p0, N1, w.xa, w.A1, w.perm0);
    run_gcn(stream, w, w.A1, N1, w.xa, 256, wd1, bd1, 256, 1, w.x1);       // xs[1]

    run_pool(stream, w, w.A1, N1, w.x1, p1, N2, w.xa, w.A2, w.perm1);
    run_gcn(stream, w, w.A2, N2, w.xa, 256, wd2, bd2, 256, 1, w.x2);       // xs[2]

    run_pool(stream, w, w.A2, N2, w.x2, p2, N3, w.xa, w.A3, w.perm2);
    run_gcn(stream, w, w.A3, N3, w.xa, 256, wd3, bd3, 256, 1, w.xb);       // bottom x (512x256)

    // ---- Up path ----
    // i=0, j=2: unpool to 1024
    hipMemcpyAsync(w.xa, w.x2, (size_t)N2 * 256 * 4, hipMemcpyDeviceToDevice, stream);
    k_scatter_add<<<gblocks((size_t)N3 * 256), THREADS, 0, stream>>>(w.xb, w.perm2, w.xa, N3, 256);
    run_gcn(stream, w, w.A2, N2, w.xa, 256, wu0, bu0, 256, 1, w.xb);

    // i=1, j=1: unpool to 2048
    hipMemcpyAsync(w.xa, w.x1, (size_t)N1 * 256 * 4, hipMemcpyDeviceToDevice, stream);
    k_scatter_add<<<gblocks((size_t)N2 * 256), THREADS, 0, stream>>>(w.xb, w.perm1, w.xa, N2, 256);
    run_gcn(stream, w, w.A1, N1, w.xa, 256, wu1, bu1, 256, 1, w.xb);

    // i=2, j=0: unpool to 4096, final GCN (no relu) -> d_out (4096 x 128 f32)
    hipMemcpyAsync(w.xa, w.x0, (size_t)N0 * 256 * 4, hipMemcpyDeviceToDevice, stream);
    k_scatter_add<<<gblocks((size_t)N1 * 256), THREADS, 0, stream>>>(w.xb, w.perm0, w.xa, N1, 256);
    run_gcn(stream, w, w.A0, N0, w.xa, 256, wu2, bu2, 128, 0, (float*)d_out);
}
